// DualRefine_72756745994445
// MI455X (gfx1250) — matrix operand order repaired and checked by reading, not compiled, not run on hardware
//
#include <hip/hip_runtime.h>
#include <hip/hip_bf16.h>

// ---------------------------------------------------------------------------
// DualRefine forward for gfx1250 (MI455X). Heavy GEMMs use
// v_wmma_f32_16x16x32_f16 (f16 operands, f32 accumulate), 32-row tiles with
// 2x B-fragment reuse per wave, async global->LDS staging when available.
// ---------------------------------------------------------------------------

typedef _Float16 h16t;
typedef __attribute__((ext_vector_type(16))) _Float16 v16h;
typedef __attribute__((ext_vector_type(8)))  float    v8f;
typedef __attribute__((ext_vector_type(4)))  int      v4i;

#define INVS 0.9999950000374997f   // 1/sqrt(1+1e-5)
#define BB 4
#define NN 2048
#define MM 2048
#define PP (BB*NN)

#if __has_builtin(__builtin_amdgcn_global_load_async_to_lds_b128) && \
    __has_builtin(__builtin_amdgcn_s_wait_asynccnt)
#define HAVE_ASYNC_LDS 1
#else
#define HAVE_ASYNC_LDS 0
#endif

// ---- WMMA fragment loaders (contiguous k order, same in A and B) ------------------
// A 16x32 f16, row-major source (ldm halves). lane L: m=L&15, hi=L>>4.
// halves 0..15 -> K = hi*16 + 0..15 (contiguous)
__device__ __forceinline__ v16h load_a16x32(const h16t* src, int ldm, int kb, int lane) {
    int m = lane & 15, hi = lane >> 4;
    union { v16h h; uint4 q[2]; } f;
    f.q[0] = *(const uint4*)(src + (long)m * ldm + kb + hi * 16);
    f.q[1] = *(const uint4*)(src + (long)m * ldm + kb + hi * 16 + 8);
    return f.h;
}
// B 32x16 f16 with B[k][n] = W[n][k], W row-major [OUT][ldw].
// lane L: n=L&15, hi=L>>4; halves 0..15 -> K = hi*16 + 0..15 (contiguous).
__device__ __forceinline__ v16h load_b32x16(const h16t* W, int ldw, int nbase, int kb, int lane) {
    int n = nbase + (lane & 15), hi = lane >> 4;
    const uint4* p = (const uint4*)(W + (long)n * ldw + kb + hi * 16);
    union { v16h h; uint4 q[2]; } f;
    f.q[0] = p[0];
    f.q[1] = p[1];
    return f.h;
}

// ---- K0: convert weights to f16 (pad K 963 -> 992 with zeros) --------------
__global__ __launch_bounds__(256) void convert_weights(
    const float* w3, const float* c1w, const float* wsw, const float* c2w, const float* h1w,
    h16t* w3h, h16t* c1h, h16t* wsh, h16t* c2h, h16t* h1h)
{
    long i0 = (long)blockIdx.x * 256 + threadIdx.x;
    long stride = (long)gridDim.x * 256;
    for (long i = i0; i < 512L * 128; i += stride) w3h[i] = (h16t)w3[i];
    for (long i = i0; i < 1024L * 992; i += stride) {
        long r = i / 992, c = i % 992;
        c1h[i] = (h16t)(c < 963 ? c1w[r * 963 + c] : 0.f);
    }
    for (long i = i0; i < 512L * 992; i += stride) {
        long r = i / 992, c = i % 992;
        wsh[i] = (h16t)(c < 963 ? wsw[r * 963 + c] : 0.f);
    }
    for (long i = i0; i < 512L * 1024; i += stride) c2h[i] = (h16t)c2w[i];
    for (long i = i0; i < 256L * 512; i += stride) h1h[i] = (h16t)h1w[i];
}

// ---- K1: top-16 nearest neighbors (coarse->partial and coarse->coarse) ----
__global__ __launch_bounds__(256) void knn_kernel(
    const float* __restrict__ coarse, const float* __restrict__ partial,
    int* __restrict__ idx_part, int* __restrict__ idx_self)
{
    int gid = blockIdx.x * 256 + threadIdx.x;
    if (gid >= PP) return;
    int b = gid / NN, n = gid % NN;
    float cx = coarse[(b * 3 + 0) * NN + n];
    float cy = coarse[(b * 3 + 1) * NN + n];
    float cz = coarse[(b * 3 + 2) * NN + n];

    float bd[16]; int bi[16];
#pragma unroll
    for (int j = 0; j < 16; ++j) { bd[j] = 3.0e38f; bi[j] = 0; }
    for (int m = 0; m < MM; ++m) {
        float dx = partial[(b * 3 + 0) * MM + m] - cx;
        float dy = partial[(b * 3 + 1) * MM + m] - cy;
        float dz = partial[(b * 3 + 2) * MM + m] - cz;
        float d = dx * dx + dy * dy + dz * dz;
        if (d < bd[15]) {
            bd[15] = d; bi[15] = m;
#pragma unroll
            for (int j = 15; j > 0; --j) {
                if (bd[j] < bd[j - 1]) {
                    float td = bd[j]; bd[j] = bd[j - 1]; bd[j - 1] = td;
                    int   ti = bi[j]; bi[j] = bi[j - 1]; bi[j - 1] = ti;
                }
            }
        }
    }
#pragma unroll
    for (int j = 0; j < 16; ++j) idx_part[(long)gid * 16 + j] = bi[j];

#pragma unroll
    for (int j = 0; j < 16; ++j) { bd[j] = 3.0e38f; bi[j] = 0; }
    for (int m = 0; m < NN; ++m) {
        float dx = coarse[(b * 3 + 0) * NN + m] - cx;
        float dy = coarse[(b * 3 + 1) * NN + m] - cy;
        float dz = coarse[(b * 3 + 2) * NN + m] - cz;
        float d = dx * dx + dy * dy + dz * dz;
        if (d < bd[15]) {
            bd[15] = d; bi[15] = m;
#pragma unroll
            for (int j = 15; j > 0; --j) {
                if (bd[j] < bd[j - 1]) {
                    float td = bd[j]; bd[j] = bd[j - 1]; bd[j - 1] = td;
                    int   ti = bi[j]; bi[j] = bi[j - 1]; bi[j - 1] = ti;
                }
            }
        }
    }
#pragma unroll
    for (int j = 0; j < 16; ++j) idx_self[(long)gid * 16 + j] = bi[j];
}

// ---- K2: partial_refine MLP; one block per point, WMMA for 128->512 -------
// 8 waves: wave w owns 64 output cols x both 16-row tiles (B reused 2x).
__global__ __launch_bounds__(256) void pr_mlp_kernel(
    const float* __restrict__ coarse, const float* __restrict__ partial,
    const int* __restrict__ idx_part, const int* __restrict__ idx_self,
    const float* __restrict__ w1, const float* __restrict__ g1, const float* __restrict__ b1,
    const float* __restrict__ w2, const float* __restrict__ g2, const float* __restrict__ b2,
    const h16t* __restrict__ w3h, const float* __restrict__ g3, const float* __restrict__ b3,
    float* __restrict__ offp)
{
    __shared__ float featS[32 * 6];
    __shared__ float h1S[32 * 32];
    __shared__ __align__(16) h16t h2S[32 * 136];   // ldm=136 (16B-aligned rows)
    __shared__ float pmax[2 * 512];

    int pt = blockIdx.x;
    int b = pt / NN, n = pt % NN;
    int tid = threadIdx.x;
    float cx = coarse[(b * 3 + 0) * NN + n];
    float cy = coarse[(b * 3 + 1) * NN + n];
    float cz = coarse[(b * 3 + 2) * NN + n];

    if (tid < 32) {
        int k = tid;
        float px, py, pz;
        if (k < 16) {        // partial neighbors first (knn)
            int m = idx_part[(long)pt * 16 + k];
            px = partial[(b * 3 + 0) * MM + m];
            py = partial[(b * 3 + 1) * MM + m];
            pz = partial[(b * 3 + 2) * MM + m];
        } else {             // then self neighbors (knn_self)
            int m = idx_self[(long)pt * 16 + (k - 16)];
            px = coarse[(b * 3 + 0) * NN + m];
            py = coarse[(b * 3 + 1) * NN + m];
            pz = coarse[(b * 3 + 2) * NN + m];
        }
        featS[k * 6 + 0] = cx; featS[k * 6 + 1] = cy; featS[k * 6 + 2] = cz;
        featS[k * 6 + 3] = px - cx; featS[k * 6 + 4] = py - cy; featS[k * 6 + 5] = pz - cz;
    }
    __syncthreads();

    // layer1: 6 -> 32, bn + leaky_relu(0.2)
    for (int i = tid; i < 32 * 32; i += 256) {
        int k = i >> 5, o = i & 31;
        float s = 0.f;
#pragma unroll
        for (int c = 0; c < 6; ++c) s += featS[k * 6 + c] * w1[o * 6 + c];
        s = s * INVS * g1[o] + b1[o];
        h1S[k * 32 + o] = s > 0.f ? s : 0.2f * s;
    }
    __syncthreads();

    // layer2: 32 -> 128, bn + leaky_relu, store f16 for WMMA A
    for (int i = tid; i < 32 * 128; i += 256) {
        int k = i >> 7, o = i & 127;
        float s = 0.f;
#pragma unroll
        for (int c = 0; c < 32; ++c) s += h1S[k * 32 + c] * w2[o * 32 + c];
        s = s * INVS * g2[o] + b2[o];
        s = s > 0.f ? s : 0.2f * s;
        h2S[k * 136 + o] = (h16t)s;
    }
    __syncthreads();

    // layer3: 128 -> 512 via WMMA.
    int lane = tid & 31, w = tid >> 5;
    int cgbase = w * 64;
    v8f acc[2][4];
#pragma unroll
    for (int rt = 0; rt < 2; ++rt)
#pragma unroll
        for (int cb = 0; cb < 4; ++cb)
#pragma unroll
            for (int r = 0; r < 8; ++r) acc[rt][cb][r] = 0.f;

#pragma unroll
    for (int kc = 0; kc < 4; ++kc) {
        int kb = kc * 32;
        v16h a0 = load_a16x32(h2S, 136, kb, lane);
        v16h a1 = load_a16x32(h2S + 16 * 136, 136, kb, lane);
#pragma unroll
        for (int cb = 0; cb < 4; ++cb) {
            v16h bf = load_b32x16(w3h, 128, cgbase + cb * 16, kb, lane);
            acc[0][cb] = __builtin_amdgcn_wmma_f32_16x16x32_f16(
                false, a0, false, bf, (short)0, acc[0][cb], false, false);
            acc[1][cb] = __builtin_amdgcn_wmma_f32_16x16x32_f16(
                false, a1, false, bf, (short)0, acc[1][cb], false, false);
        }
    }

    // bn + leaky_relu + max over rows of each 16-row tile
    int nl = lane & 15, hiq = lane >> 4;
#pragma unroll
    for (int rt = 0; rt < 2; ++rt) {
#pragma unroll
        for (int cb = 0; cb < 4; ++cb) {
            int o = cgbase + cb * 16 + nl;
            float gg = g3[o] * INVS, bbv = b3[o];
            float m8 = -3.0e38f;
#pragma unroll
            for (int r = 0; r < 8; ++r) {
                float v = acc[rt][cb][r] * gg + bbv;
                v = v > 0.f ? v : 0.2f * v;
                m8 = fmaxf(m8, v);
            }
            m8 = fmaxf(m8, __shfl_xor(m8, 16, 32));   // combine lane halves (wave32)
            if (hiq == 0) pmax[rt * 512 + o] = m8;
        }
    }
    __syncthreads();
    for (int c = tid; c < 512; c += 256)
        offp[(long)pt * 512 + c] = fmaxf(pmax[c], pmax[512 + c]);
}

// ---- K3: projection (bilinear sampling) + assemble z as f16 [P][992] ------
__global__ __launch_bounds__(256) void proj_kernel(
    const float* __restrict__ coarse,
    const float* __restrict__ f0, const float* __restrict__ f1,
    const float* __restrict__ f2, const float* __restrict__ f3,
    h16t* __restrict__ zh)
{
    int pt = blockIdx.x;
    int b = pt / NN, n = pt % NN;
    float cx = coarse[(b * 3 + 0) * NN + n];
    float cy = coarse[(b * 3 + 1) * NN + n];
    float cz = coarse[(b * 3 + 2) * NN + n];
    float hh = fminf(fmaxf(248.f * cy / cz + 111.5f, 0.f), 223.f);
    float ww = fminf(fmaxf(248.f * cx / (-cz) + 111.5f, 0.f), 223.f);

    for (int j = threadIdx.x; j < 992; j += 256) {
        float v;
        if (j < 3) {
            v = (j == 0) ? cx : ((j == 1) ? cy : cz);
        } else if (j < 963) {
            int c = j - 3;
            const float* img; int S, C, ch;
            if (c < 64)       { img = f0; S = 56; C = 64;  ch = c; }
            else if (c < 192) { img = f1; S = 28; C = 128; ch = c - 64; }
            else if (c < 448) { img = f2; S = 14; C = 256; ch = c - 192; }
            else              { img = f3; S = 7;  C = 512; ch = c - 448; }
            float x = hh / (224.f / (float)S);
            float y = ww / (224.f / (float)S);
            int x1 = (int)floorf(x), y1 = (int)floorf(y);
            int x2 = min((int)ceilf(x), S - 1), y2 = min((int)ceilf(y), S - 1);
            int xl = (int)x, yl = (int)y;
            float w11 = (float)((x2 - xl) * (y2 - yl));
            float w12 = (float)((x2 - xl) * (yl - y1));
            float w21 = (float)((xl - x1) * (y2 - yl));
            float w22 = (float)((xl - x1) * (yl - y1));
            const float* base = img + ((long)b * C + ch) * S * S;
            v = w11 * base[x1 * S + y1] + w21 * base[x2 * S + y1]
              + w12 * base[x1 * S + y2] + w22 * base[x2 * S + y2];
        } else {
            v = 0.f;  // K padding 963 -> 992
        }
        zh[(long)pt * 992 + j] = (h16t)v;
    }
}

// ---- generic WMMA GEMM: 32-point row tile per block -----------------------
// Each wave owns CPW cols x both 16-row tiles, so every B fragment feeds two
// v_wmma ops. A staged in LDS via async global->LDS loads when available.
// EPI 0: (acc+bias)*INVS*g + beta, relu -> outH (f16)           [ir_c1]
// EPI 1: outF = addsrc + acc + bias                              [ir_ws + off_p]
// EPI 2: t = addsrc + acc + bias -> outF (f32) and outH (f16)    [ir_c2]
// EPI 3: (acc+bias)*INVS*g + beta, relu -> outF (f32)            [h1]
template <int KW, int OUT, int EPI, int NWAVES>
__global__ __launch_bounds__(NWAVES * 32) void gemm32_kernel(
    const h16t* __restrict__ A, const h16t* __restrict__ W,
    const float* __restrict__ bias, const float* __restrict__ gamma,
    const float* __restrict__ beta, const float* __restrict__ addsrc,
    float* __restrict__ outF, h16t* __restrict__ outH)
{
    constexpr int THREADS = NWAVES * 32;
    constexpr int CPW = OUT / NWAVES;          // cols per wave
    constexpr int NCB = CPW / 16;              // 16-col blocks per wave
    __shared__ __align__(16) h16t ldsA[32 * KW];
    int tid = threadIdx.x, lane = tid & 31, w = tid >> 5;
    long rowbase = (long)blockIdx.x * 32;

    // stage A tile (32 x KW f16) into LDS
#if HAVE_ASYNC_LDS
    {
        const char* gsrc = (const char*)(A + rowbase * KW);
        for (int i = tid; i < 4 * KW; i += THREADS) {  // 16B chunks
            __builtin_amdgcn_global_load_async_to_lds_b128(
                (__attribute__((address_space(1))) v4i*)(gsrc + (long)i * 16),
                (__attribute__((address_space(3))) v4i*)((char*)ldsA + (long)i * 16),
                0, 0);
        }
        __builtin_amdgcn_s_wait_asynccnt(0);
    }
#else
    {
        const unsigned* src = (const unsigned*)(A + rowbase * KW);
        unsigned* dst = (unsigned*)ldsA;
        for (int i = tid; i < 16 * KW; i += THREADS) dst[i] = src[i];
    }
#endif
    __syncthreads();

    v8f acc[2][NCB];
#pragma unroll
    for (int rt = 0; rt < 2; ++rt)
#pragma unroll
        for (int cb = 0; cb < NCB; ++cb)
#pragma unroll
            for (int r = 0; r < 8; ++r) acc[rt][cb][r] = 0.f;

    int colbase = w * CPW;
    __builtin_prefetch(W + (long)colbase * KW, 0, 3);

    for (int kc = 0; kc < KW / 32; ++kc) {
        int kb = kc * 32;
        v16h a0 = load_a16x32(ldsA, KW, kb, lane);
        v16h a1 = load_a16x32(ldsA + 16 * KW, KW, kb, lane);
#pragma unroll
        for (int cb = 0; cb < NCB; ++cb) {
            v16h bf = load_b32x16(W, KW, colbase + cb * 16, kb, lane);
            acc[0][cb] = __builtin_amdgcn_wmma_f32_16x16x32_f16(
                false, a0, false, bf, (short)0, acc[0][cb], false, false);
            acc[1][cb] = __builtin_amdgcn_wmma_f32_16x16x32_f16(
                false, a1, false, bf, (short)0, acc[1][cb], false, false);
        }
    }

    int nl = lane & 15, hi = lane >> 4;
#pragma unroll
    for (int rt = 0; rt < 2; ++rt) {
#pragma unroll
        for (int cb = 0; cb < NCB; ++cb) {
            int nc = colbase + cb * 16 + nl;
#pragma unroll
            for (int r = 0; r < 8; ++r) {
                long pt = rowbase + rt * 16 + r + hi * 8;
                float v = acc[rt][cb][r];
                if (EPI == 0) {
                    v = (v + bias[nc]) * INVS * gamma[nc] + beta[nc];
                    v = v > 0.f ? v : 0.f;
                    outH[pt * OUT + nc] = (h16t)v;
                } else if (EPI == 1) {
                    outF[pt * OUT + nc] = addsrc[pt * OUT + nc] + v + bias[nc];
                } else if (EPI == 2) {
                    float t = addsrc[pt * OUT + nc] + v + bias[nc];
                    outF[pt * OUT + nc] = t;
                    outH[pt * OUT + nc] = (h16t)t;
                } else {
                    v = (v + bias[nc]) * INVS * gamma[nc] + beta[nc];
                    v = v > 0.f ? v : 0.f;
                    outF[pt * OUT + nc] = v;
                }
            }
        }
    }
}

// ---- K6: head 256 -> 128 -> 32 -> 3 (+tanh), one block per point ----------
__global__ __launch_bounds__(256) void head_kernel(
    const float* __restrict__ h1out,
    const float* __restrict__ w2, const float* __restrict__ b2,
    const float* __restrict__ g2, const float* __restrict__ bb2,
    const float* __restrict__ w3, const float* __restrict__ b3,
    const float* __restrict__ g3, const float* __restrict__ bb3,
    const float* __restrict__ w4, const float* __restrict__ b4,
    float* __restrict__ out)
{
    __shared__ float aS[256];
    __shared__ float bS[128];
    __shared__ float cS[32];
    int pt = blockIdx.x, t = threadIdx.x;
    aS[t] = h1out[(long)pt * 256 + t];
    __syncthreads();
    if (t < 128) {
        float s = 0.f;
        for (int c = 0; c < 256; ++c) s += aS[c] * w2[t * 256 + c];
        s = (s + b2[t]) * INVS * g2[t] + bb2[t];
        bS[t] = s > 0.f ? s : 0.f;
    }
    __syncthreads();
    if (t < 32) {
        float s = 0.f;
        for (int c = 0; c < 128; ++c) s += bS[c] * w3[t * 128 + c];
        s = (s + b3[t]) * INVS * g3[t] + bb3[t];
        cS[t] = s > 0.f ? s : 0.f;
    }
    __syncthreads();
    if (t < 3) {
        float s = b4[t];
        for (int c = 0; c < 32; ++c) s += cS[c] * w4[t * 32 + c];
        out[(long)pt * 3 + t] = tanhf(s);   // output (B, N, 3) row-major
    }
}

// ---------------------------------------------------------------------------
extern "C" void kernel_launch(void* const* d_in, const int* in_sizes, int n_in,
                              void* d_out, int out_size, void* d_ws, size_t ws_size,
                              hipStream_t stream) {
    const float* coarse  = (const float*)d_in[0];
    const float* partial = (const float*)d_in[1];
    const float* img_f0  = (const float*)d_in[2];
    const float* img_f1  = (const float*)d_in[3];
    const float* img_f2  = (const float*)d_in[4];
    const float* img_f3  = (const float*)d_in[5];
    const float* pr_w1 = (const float*)d_in[6];
    const float* pr_g1 = (const float*)d_in[7];
    const float* pr_b1 = (const float*)d_in[8];
    const float* pr_w2 = (const float*)d_in[9];
    const float* pr_g2 = (const float*)d_in[10];
    const float* pr_b2 = (const float*)d_in[11];
    const float* pr_w3 = (const float*)d_in[12];
    const float* pr_g3 = (const float*)d_in[13];
    const float* pr_b3 = (const float*)d_in[14];
    const float* ir_ws_w = (const float*)d_in[15];
    const float* ir_ws_b = (const float*)d_in[16];
    const float* ir_c1_w = (const float*)d_in[17];
    const float* ir_c1_b = (const float*)d_in[18];
    const float* ir_g1   = (const float*)d_in[19];
    const float* ir_b1   = (const float*)d_in[20];
    const float* ir_c2_w = (const float*)d_in[21];
    const float* ir_c2_b = (const float*)d_in[22];
    const float* h1_w  = (const float*)d_in[23];
    const float* h1_b  = (const float*)d_in[24];
    const float* h1_g  = (const float*)d_in[25];
    const float* h1_bb = (const float*)d_in[26];
    const float* h2_w  = (const float*)d_in[27];
    const float* h2_b  = (const float*)d_in[28];
    const float* h2_g  = (const float*)d_in[29];
    const float* h2_bb = (const float*)d_in[30];
    const float* h3_w  = (const float*)d_in[31];
    const float* h3_b  = (const float*)d_in[32];
    const float* h3_g  = (const float*)d_in[33];
    const float* h3_bb = (const float*)d_in[34];
    const float* h4_w  = (const float*)d_in[35];
    const float* h4_b  = (const float*)d_in[36];

    char* ws = (char*)d_ws;
    size_t o_idxp = 0;
    size_t o_idxs = o_idxp + (size_t)PP * 16 * 4;
    size_t o_w3h  = o_idxs + (size_t)PP * 16 * 4;
    size_t o_c1h  = o_w3h + 512L * 128 * 2;
    size_t o_wsh  = o_c1h + 1024L * 992 * 2;
    size_t o_c2h  = o_wsh + 512L * 992 * 2;
    size_t o_h1h  = o_c2h + 512L * 1024 * 2;
    size_t o_zh   = o_h1h + 256L * 512 * 2;
    size_t o_oh   = o_zh  + (size_t)PP * 992 * 2;
    size_t o_offp = o_oh  + (size_t)PP * 1024 * 2;
    size_t o_xbuf = o_offp + (size_t)PP * 512 * 4;
    size_t o_xh   = o_xbuf + (size_t)PP * 512 * 4;
    size_t o_h1o  = o_xh   + (size_t)PP * 512 * 2;

    int*  idx_part = (int*)(ws + o_idxp);
    int*  idx_self = (int*)(ws + o_idxs);
    h16t* w3h  = (h16t*)(ws + o_w3h);
    h16t* c1h  = (h16t*)(ws + o_c1h);
    h16t* wsh  = (h16t*)(ws + o_wsh);
    h16t* c2h  = (h16t*)(ws + o_c2h);
    h16t* h1h  = (h16t*)(ws + o_h1h);
    h16t* zh   = (h16t*)(ws + o_zh);
    h16t* oh   = (h16t*)(ws + o_oh);
    float* offp = (float*)(ws + o_offp);
    float* xbuf = (float*)(ws + o_xbuf);
    h16t*  xh   = (h16t*)(ws + o_xh);
    float* h1o  = (float*)(ws + o_h1o);
    float* out  = (float*)d_out;

    convert_weights<<<1024, 256, 0, stream>>>(pr_w3, ir_c1_w, ir_ws_w, ir_c2_w, h1_w,
                                              w3h, c1h, wsh, c2h, h1h);
    knn_kernel<<<PP / 256, 256, 0, stream>>>(coarse, partial, idx_part, idx_self);
    pr_mlp_kernel<<<PP, 256, 0, stream>>>(coarse, partial, idx_part, idx_self,
                                          pr_w1, pr_g1, pr_b1, pr_w2, pr_g2, pr_b2,
                                          w3h, pr_g3, pr_b3, offp);
    proj_kernel<<<PP, 256, 0, stream>>>(coarse, img_f0, img_f1, img_f2, img_f3, zh);
    // ir_c1: z(992) -> 1024, bn+relu -> o (f16). 16 waves/block.
    gemm32_kernel<992, 1024, 0, 16><<<PP / 32, 512, 0, stream>>>(
        zh, c1h, ir_c1_b, ir_g1, ir_b1, nullptr, nullptr, oh);
    // ir_ws: z(992) -> 512, + ws_b + off_p -> xbuf
    gemm32_kernel<992, 512, 1, 8><<<PP / 32, 256, 0, stream>>>(
        zh, wsh, ir_ws_b, nullptr, nullptr, offp, xbuf, nullptr);
    // ir_c2: o(1024) -> 512, + c2_b + xbuf -> xbuf (f32) and xh (f16)
    gemm32_kernel<1024, 512, 2, 8><<<PP / 32, 256, 0, stream>>>(
        oh, c2h, ir_c2_b, nullptr, nullptr, xbuf, xbuf, xh);
    // h1: x(512) -> 256, bn+relu -> h1o (f32)
    gemm32_kernel<512, 256, 3, 8><<<PP / 32, 256, 0, stream>>>(
        xh, h1h, h1_b, h1_g, h1_bb, nullptr, h1o, nullptr);
    head_kernel<<<PP, 256, 0, stream>>>(h1o, h2_w, h2_b, h2_g, h2_bb,
                                        h3_w, h3_b, h3_g, h3_bb, h4_w, h4_b, out);
}
